// LoRCnnAttention_68118181314885
// MI455X (gfx1250) — compile-verified, matrix operand order checked by reading, not run on hardware
//
#include <hip/hip_runtime.h>
#include <cstdint>
#include <cstring>

// ---------------- problem constants ----------------
constexpr int S_   = 1024;
constexpr int HID_ = 4096;
constexpr int H_   = 32;
constexpr int DH_  = 128;
constexpr int DL_  = 64;
constexpr int KC_  = 63;     // conv taps
constexpr int NL_  = 3;      // conv layers
constexpr float EPS_ = 1e-5f;

// ---------------- types ----------------
typedef __attribute__((ext_vector_type(8)))  __bf16 v8bf;
typedef __attribute__((ext_vector_type(16))) __bf16 v16bf;
typedef __attribute__((ext_vector_type(8)))  float  v8f;
typedef __attribute__((ext_vector_type(4)))  unsigned int v4u;
typedef __attribute__((ext_vector_type(8)))  int v8i_;
typedef __attribute__((ext_vector_type(4)))  int v4i_;
typedef unsigned short u16;

#if defined(__has_builtin)
#if __has_builtin(__builtin_amdgcn_tensor_load_to_lds)
#define HAVE_TDM 1
#endif
#endif

// float -> bf16 (round-to-nearest-even)
__device__ __forceinline__ u16 f2b(float x) {
    unsigned int u = __float_as_uint(x);
    unsigned int r = (u + 0x7FFFu + ((u >> 16) & 1u)) >> 16;
    return (u16)r;
}

// ---------------- convert kernel ----------------
__global__ void f32_to_bf16_kernel(const float* __restrict__ src,
                                   u16* __restrict__ dst, int n) {
    int i = blockIdx.x * blockDim.x + threadIdx.x;
    int stride = gridDim.x * blockDim.x;
    for (; i < n; i += stride) dst[i] = f2b(src[i]);
}

// ---------------- WMMA bf16 GEMM:  C = alpha * A(MxK) * B(NxK)^T ----------------
// A row-major [M,K] (lda), B row-major [N,K] (ldb), C row-major [M,N] (ldc).
// Block = 256 threads = 8 waves; wave tile 32x32 (2x2 wmma), block tile 128x64.
// Grid: (N/64, M/128, batch).

__device__ __forceinline__ v16bf load_frag(const u16* __restrict__ p, int ld,
                                           int row0, int k0, int lane) {
    int m = lane & 15;
    int h = lane >> 4;
    const u16* base = p + (size_t)(row0 + m) * (size_t)ld + (size_t)(k0 + 8 * h);
    v8bf lo = *(const v8bf*)(const void*)(base);       // K = 8h .. 8h+7
    v8bf hi = *(const v8bf*)(const void*)(base + 16);  // K = 8h+16 .. 8h+23
    return __builtin_shufflevector(lo, hi, 0, 1, 2, 3, 4, 5, 6, 7,
                                   8, 9, 10, 11, 12, 13, 14, 15);
}

__device__ __forceinline__ void store_tile(float* __restrict__ C, int ldc,
                                           int row, int col, v8f acc,
                                           float alpha, int lane) {
    int n  = lane & 15;
    int hh = lane >> 4;
#pragma unroll
    for (int r = 0; r < 8; ++r)
        C[(size_t)(row + r + 8 * hh) * (size_t)ldc + (size_t)(col + n)] = alpha * acc[r];
}

__global__ void gemm_bf16_wmma(const u16* __restrict__ A,
                               const u16* __restrict__ B,
                               float* __restrict__ C,
                               int Kdim, int lda, int ldb, int ldc,
                               long long sA, long long sB, long long sC,
                               float alpha) {
    const int lane = threadIdx.x & 31;
    const int wave = threadIdx.x >> 5;
    const int wy = wave >> 1;   // 0..3
    const int wx = wave & 1;    // 0..1
    const int row0 = blockIdx.y * 128 + wy * 32;
    const int col0 = blockIdx.x * 64  + wx * 32;

    const u16* Ab = A + (size_t)blockIdx.z * (size_t)sA;
    const u16* Bb = B + (size_t)blockIdx.z * (size_t)sB;
    float*     Cb = C + (size_t)blockIdx.z * (size_t)sC;

    v8f acc00 = {}, acc01 = {}, acc10 = {}, acc11 = {};

    for (int kt = 0; kt < Kdim; kt += 32) {
        if (kt + 64 < Kdim) {
            // speculative stream-ahead -> global_prefetch_b8
            __builtin_prefetch(Ab + (size_t)(row0 + (lane & 15)) * lda + kt + 64, 0, 1);
            __builtin_prefetch(Bb + (size_t)(col0 + (lane & 15)) * ldb + kt + 64, 0, 1);
        }
        v16bf a0 = load_frag(Ab, lda, row0,      kt, lane);
        v16bf a1 = load_frag(Ab, lda, row0 + 16, kt, lane);
        v16bf b0 = load_frag(Bb, ldb, col0,      kt, lane);
        v16bf b1 = load_frag(Bb, ldb, col0 + 16, kt, lane);
        acc00 = __builtin_amdgcn_wmma_f32_16x16x32_bf16(false, a0, false, b0, (short)0, acc00, false, false);
        acc01 = __builtin_amdgcn_wmma_f32_16x16x32_bf16(false, a0, false, b1, (short)0, acc01, false, false);
        acc10 = __builtin_amdgcn_wmma_f32_16x16x32_bf16(false, a1, false, b0, (short)0, acc10, false, false);
        acc11 = __builtin_amdgcn_wmma_f32_16x16x32_bf16(false, a1, false, b1, (short)0, acc11, false, false);
    }

    store_tile(Cb, ldc, row0,      col0,      acc00, alpha, lane);
    store_tile(Cb, ldc, row0,      col0 + 16, acc01, alpha, lane);
    store_tile(Cb, ldc, row0 + 16, col0,      acc10, alpha, lane);
    store_tile(Cb, ldc, row0 + 16, col0 + 16, acc11, alpha, lane);
}

// ---------------- RoPE + low-rank down-proj + v pack ----------------
// grid (S, H), block 128.  qf/kf/vf are [S, HID] f32.
// Outputs: qlb/klb [H, S, DL] bf16;  vbt [H, DH, S] bf16 (transposed for A*B^T pv GEMM).
__global__ void rope_down_pack(const float* __restrict__ qf,
                               const float* __restrict__ kf,
                               const float* __restrict__ vf,
                               const float* __restrict__ Wdq,
                               const float* __restrict__ Wdk,
                               const int*   __restrict__ pos,
                               u16* __restrict__ qlb,
                               u16* __restrict__ klb,
                               u16* __restrict__ vbt) {
    const int s = blockIdx.x;
    const int h = blockIdx.y;
    const int d = threadIdx.x;  // 0..127
    __shared__ float sq[DH_], sk[DH_], rq[DH_], rk[DH_];

    size_t idx = (size_t)s * HID_ + (size_t)h * DH_ + d;
    float qv = qf[idx], kv = kf[idx], vv = vf[idx];
    sq[d] = qv; sk[d] = kv;
    __syncthreads();

    int f = d & 63;
    float inv = powf(10000.0f, -(float)(2 * f) / 128.0f);
    float ang = (float)pos[s] * inv;
    float c = cosf(ang), sn = sinf(ang);
    float oq = (d < 64) ? -sq[d + 64] : sq[d - 64];
    float ok = (d < 64) ? -sk[d + 64] : sk[d - 64];
    rq[d] = qv * c + oq * sn;
    rk[d] = kv * c + ok * sn;
    __syncthreads();

    if (d < DL_) {
        float aq = 0.f, ak = 0.f;
#pragma unroll 8
        for (int t = 0; t < DH_; ++t) {
            aq += rq[t] * Wdq[d * DH_ + t];
            ak += rk[t] * Wdk[d * DH_ + t];
        }
        size_t o = ((size_t)h * S_ + s) * DL_ + d;
        qlb[o] = f2b(aq);
        klb[o] = f2b(ak);
    }
    vbt[((size_t)h * DH_ + d) * S_ + s] = f2b(vv);
}

// ---------------- per-row stats: LN1 mean/rstd + est_scale ----------------
// grid (H*S), block 256. attn [H*S, S] f32.
__global__ void row_stats_est(const float* __restrict__ attn,
                              const float* __restrict__ scw,
                              const float* __restrict__ scb,
                              float* __restrict__ meanb,
                              float* __restrict__ rstdb,
                              float* __restrict__ estb) {
    const int row = blockIdx.x;
    const float* x = attn + (size_t)row * S_;
    __shared__ float r0[256], r1[256], r2[256];
    float s = 0.f, s2 = 0.f, e = 0.f;
#pragma unroll
    for (int j = 0; j < 4; ++j) {
        int k = threadIdx.x + j * 256;
        float v = x[k];
        s += v; s2 += v * v; e += v * scw[k];
    }
    r0[threadIdx.x] = s; r1[threadIdx.x] = s2; r2[threadIdx.x] = e;
    __syncthreads();
    for (int off = 128; off > 0; off >>= 1) {
        if (threadIdx.x < off) {
            r0[threadIdx.x] += r0[threadIdx.x + off];
            r1[threadIdx.x] += r1[threadIdx.x + off];
            r2[threadIdx.x] += r2[threadIdx.x + off];
        }
        __syncthreads();
    }
    if (threadIdx.x == 0) {
        float m = r0[0] / (float)S_;
        float var = r1[0] / (float)S_ - m * m;
        meanb[row] = m;
        rstdb[row] = rsqrtf(var + EPS_);
        estb[row]  = r2[0] + scb[0];
    }
}

// ---------------- fused LN1 + 3x causal depthwise conv (in LDS ping-pong) ----------------
// grid (S/KT, H), block 256, dynamic LDS = 2*1024*LST floats.
// Tile ingest uses the Tensor Data Mover: one tensor_load_to_lds DMAs the
// [1024 x 16]-f32 tile (4KB row pitch) into LDS with 1-dword padding every
// 16 dwords (== LST_=17 bank-conflict-free layout), tracked by TENSORcnt.
constexpr int KT_  = 16;   // key columns per block
constexpr int LST_ = 17;   // padded LDS row stride
__global__ void cnn_kernel(float* __restrict__ attn,
                           const float* __restrict__ convw,
                           const float* __restrict__ convb,
                           const float* __restrict__ meanb,
                           const float* __restrict__ rstdb,
                           const float* __restrict__ ln1w,
                           const float* __restrict__ ln1b) {
    extern __shared__ float sm[];
    float* bufA = sm;
    float* bufB = sm + (size_t)S_ * LST_;

    const int h  = blockIdx.y;
    const int k0 = blockIdx.x * KT_;
    float* base = attn + (size_t)h * S_ * S_;

#if defined(HAVE_TDM)
    // ---- TDM tile load: global [1024 x 16] f32 (stride 1024) -> LDS (stride 17) ----
    if (threadIdx.x == 0) {
        const float* gsrc = base + k0;                     // element [q=0, k=k0]
        unsigned long long ga = (unsigned long long)(uintptr_t)gsrc;
        unsigned lds_off = (unsigned)(uintptr_t)(void*)bufA;  // LDS aperture: low 32 bits = LDS byte offset

        // D# group 0: count=1 | lds_addr | global_addr[56:0] | type=2
        v4u g0 = { 1u,
                   lds_off,
                   (unsigned)ga,
                   (unsigned)((ga >> 32) & 0x01FFFFFFull) | 0x80000000u };

        // D# group 1:
        //  w0: data_size=2(4B)<<16 | pad_enable<<20 | pad_interval=3(16dw)<<22 | pad_amount=0(1dw)<<25
        //  tensor_dim0=1024, tensor_dim1=H*S=32768, tile_dim0=16, tile_dim1=1024,
        //  tensor_dim0_stride=1024
        v8i_ g1 = { (int)0x00D20000,
                    (int)((1024u & 0xFFFFu) << 16),                 // tdim0[15:0] @ bits63:48
                    (int)(((32768u & 0xFFFFu) << 16) | (1024u >> 16)), // tdim1[15:0] | tdim0[31:16]
                    (int)((16u << 16) | (32768u >> 16)),            // tile_dim0 | tdim1[31:16]
                    (int)1024,                                      // tile_dim1 (tile_dim2=0)
                    (int)1024,                                      // tensor_dim0_stride[31:0]
                    0, 0 };
        v4i_ gz = { 0, 0, 0, 0 };
#if __clang_major__ >= 23
        v8i_ gz8 = { 0, 0, 0, 0, 0, 0, 0, 0 };
        __builtin_amdgcn_tensor_load_to_lds(g0, g1, gz, gz, gz8, 0);
#else
        __builtin_amdgcn_tensor_load_to_lds(g0, g1, gz, gz, 0);
#endif
        __builtin_amdgcn_s_wait_tensorcnt(0);
    }
    __syncthreads();
    // apply LN1 in place in LDS
    for (int idx = threadIdx.x; idx < S_ * KT_; idx += 256) {
        int q = idx / KT_, c = idx % KT_;
        int k = k0 + c;
        float x = bufA[q * LST_ + c];
        float m = meanb[h * S_ + q], r = rstdb[h * S_ + q];
        bufA[q * LST_ + c] = (x - m) * r * ln1w[k] + ln1b[k];
    }
#else
    // fallback: manual tile load + LN1
    for (int idx = threadIdx.x; idx < S_ * KT_; idx += 256) {
        int q = idx / KT_, c = idx % KT_;
        int k = k0 + c;
        float x = base[(size_t)q * S_ + k];
        float m = meanb[h * S_ + q], r = rstdb[h * S_ + q];
        bufA[q * LST_ + c] = (x - m) * r * ln1w[k] + ln1b[k];
    }
#endif
    __syncthreads();

    float* in  = bufA;
    float* out = bufB;
    const int c    = threadIdx.x % KT_;  // column within tile
    const int qrow = threadIdx.x / KT_;  // 0..15

    for (int layer = 0; layer < NL_; ++layer) {
        const float* w = convw + (size_t)(layer * H_ + h) * KC_;
        float bias = convb[layer * H_ + h];
        float wl[KC_];
#pragma unroll
        for (int t = 0; t < KC_; ++t) wl[t] = w[t];

#pragma unroll 1
        for (int qb = 0; qb < S_; qb += 16) {
            int q = qb + qrow;
            float acc = bias;
#pragma unroll
            for (int t = 0; t < KC_; ++t) {
                int qi = q - (KC_ - 1) + t;
                if (qi >= 0) acc += wl[t] * in[qi * LST_ + c];
            }
            out[q * LST_ + c] = fmaxf(acc, 0.0f);
        }
        __syncthreads();
        float* tmp = in; in = out; out = tmp;
    }

    // result is in `in`; write back in place
    for (int idx = threadIdx.x; idx < S_ * KT_; idx += 256) {
        int q = idx / KT_, cc = idx % KT_;
        base[(size_t)q * S_ + k0 + cc] = in[q * LST_ + cc];
    }
}

// ---------------- fused LN2 + mask + softmax + sigmoid scale -> bf16 p ----------------
// grid (H*S), block 256.
__global__ void softmax_kernel(const float* __restrict__ attn,
                               const float* __restrict__ mask,
                               const float* __restrict__ ln2w,
                               const float* __restrict__ ln2b,
                               const float* __restrict__ est,
                               u16* __restrict__ pb) {
    const int row = blockIdx.x;
    const int q   = row & (S_ - 1);
    const float* x = attn + (size_t)row * S_;
    __shared__ float r0[256], r1[256];
    float v[4];

    float s = 0.f, s2 = 0.f;
#pragma unroll
    for (int j = 0; j < 4; ++j) {
        int k = threadIdx.x + j * 256;
        v[j] = x[k];
        s += v[j]; s2 += v[j] * v[j];
    }
    r0[threadIdx.x] = s; r1[threadIdx.x] = s2;
    __syncthreads();
    for (int off = 128; off > 0; off >>= 1) {
        if (threadIdx.x < off) {
            r0[threadIdx.x] += r0[threadIdx.x + off];
            r1[threadIdx.x] += r1[threadIdx.x + off];
        }
        __syncthreads();
    }
    float m   = r0[0] / (float)S_;
    float var = r1[0] / (float)S_ - m * m;
    float rstd = rsqrtf(var + EPS_);
    __syncthreads();

    float mx = -3.4028234663852886e38f;
#pragma unroll
    for (int j = 0; j < 4; ++j) {
        int k = threadIdx.x + j * 256;
        float y = (v[j] - m) * rstd * ln2w[k] + ln2b[k] + mask[(size_t)q * S_ + k];
        y = fmaxf(y, -3.4028234663852886e38f);
        v[j] = y;
        mx = fmaxf(mx, y);
    }
    r0[threadIdx.x] = mx;
    __syncthreads();
    for (int off = 128; off > 0; off >>= 1) {
        if (threadIdx.x < off)
            r0[threadIdx.x] = fmaxf(r0[threadIdx.x], r0[threadIdx.x + off]);
        __syncthreads();
    }
    float rowmax = r0[0];
    __syncthreads();

    float se = 0.f;
#pragma unroll
    for (int j = 0; j < 4; ++j) {
        float e = __expf(v[j] - rowmax);
        v[j] = e;
        se += e;
    }
    r0[threadIdx.x] = se;
    __syncthreads();
    for (int off = 128; off > 0; off >>= 1) {
        if (threadIdx.x < off) r0[threadIdx.x] += r0[threadIdx.x + off];
        __syncthreads();
    }
    float total = r0[0];
    float sig   = 1.0f / (1.0f + __expf(-est[row]));
    float scale = sig / total;
#pragma unroll
    for (int j = 0; j < 4; ++j) {
        int k = threadIdx.x + j * 256;
        pb[(size_t)row * S_ + k] = f2b(v[j] * scale);
    }
}

// ---------------- host launcher ----------------
extern "C" void kernel_launch(void* const* d_in, const int* in_sizes, int n_in,
                              void* d_out, int out_size, void* d_ws, size_t ws_size,
                              hipStream_t stream) {
    (void)in_sizes; (void)n_in; (void)out_size; (void)ws_size;

    const float* X     = (const float*)d_in[0];
    const float* mask  = (const float*)d_in[1];
    const int*   pos   = (const int*)  d_in[2];
    const float* Wq    = (const float*)d_in[3];
    const float* Wk    = (const float*)d_in[4];
    const float* Wv    = (const float*)d_in[5];
    const float* Wo    = (const float*)d_in[6];
    const float* Wdq   = (const float*)d_in[7];
    const float* Wdk   = (const float*)d_in[8];
    const float* ln1w  = (const float*)d_in[9];
    const float* ln1b  = (const float*)d_in[10];
    const float* ln2w  = (const float*)d_in[11];
    const float* ln2b  = (const float*)d_in[12];
    const float* convw = (const float*)d_in[13];
    const float* convb = (const float*)d_in[14];
    const float* scw   = (const float*)d_in[15];
    const float* scb   = (const float*)d_in[16];
    float* out = (float*)d_out;

    // workspace layout
    char* w = (char*)d_ws;
    size_t off = 0;
    auto alloc = [&](size_t bytes) -> char* {
        char* p = w + off;
        off = (off + bytes + 255) & ~(size_t)255;
        return p;
    };
    const size_t nXH = (size_t)S_ * HID_;        // 4M
    const size_t nWW = (size_t)HID_ * HID_;      // 16M
    u16*   Xb   = (u16*)  alloc(nXH * 2);
    u16*   Wqb  = (u16*)  alloc(nWW * 2);
    u16*   Wkb  = (u16*)  alloc(nWW * 2);
    u16*   Wvb  = (u16*)  alloc(nWW * 2);
    u16*   Wob  = (u16*)  alloc(nWW * 2);
    float* qf   = (float*)alloc(nXH * 4);
    float* kf   = (float*)alloc(nXH * 4);
    float* vf   = (float*)alloc(nXH * 4);
    u16*   qlb  = (u16*)  alloc((size_t)H_ * S_ * DL_ * 2);
    u16*   klb  = (u16*)  alloc((size_t)H_ * S_ * DL_ * 2);
    u16*   vbt  = (u16*)  alloc((size_t)H_ * DH_ * S_ * 2);
    float* attn = (float*)alloc((size_t)H_ * S_ * S_ * 4);
    float* mnb  = (float*)alloc((size_t)H_ * S_ * 4);
    float* rsb  = (float*)alloc((size_t)H_ * S_ * 4);
    float* estb = (float*)alloc((size_t)H_ * S_ * 4);
    u16*   pb   = (u16*)  alloc((size_t)H_ * S_ * S_ * 2);
    float* ctx  = (float*)alloc(nXH * 4);
    u16*   ctxb = (u16*)  alloc(nXH * 2);

    // 1) convert activations + weights to bf16
    f32_to_bf16_kernel<<<2048, 256, 0, stream>>>(X,  Xb,  (int)nXH);
    f32_to_bf16_kernel<<<4096, 256, 0, stream>>>(Wq, Wqb, (int)nWW);
    f32_to_bf16_kernel<<<4096, 256, 0, stream>>>(Wk, Wkb, (int)nWW);
    f32_to_bf16_kernel<<<4096, 256, 0, stream>>>(Wv, Wvb, (int)nWW);
    f32_to_bf16_kernel<<<4096, 256, 0, stream>>>(Wo, Wob, (int)nWW);

    // 2) Q/K/V projections: [1024,4096] x [4096,4096]^T
    {
        dim3 g(HID_ / 64, S_ / 128, 1), b(256);
        gemm_bf16_wmma<<<g, b, 0, stream>>>(Xb, Wqb, qf, HID_, HID_, HID_, HID_, 0, 0, 0, 1.0f);
        gemm_bf16_wmma<<<g, b, 0, stream>>>(Xb, Wkb, kf, HID_, HID_, HID_, HID_, 0, 0, 0, 1.0f);
        gemm_bf16_wmma<<<g, b, 0, stream>>>(Xb, Wvb, vf, HID_, HID_, HID_, HID_, 0, 0, 0, 1.0f);
    }

    // 3) RoPE + low-rank down-proj + v pack
    rope_down_pack<<<dim3(S_, H_), 128, 0, stream>>>(qf, kf, vf, Wdq, Wdk, pos, qlb, klb, vbt);

    // 4) attn scores: per head [1024,64] x [1024,64]^T, alpha = 1/sqrt(64)
    {
        dim3 g(S_ / 64, S_ / 128, H_), b(256);
        gemm_bf16_wmma<<<g, b, 0, stream>>>(qlb, klb, attn, DL_, DL_, DL_, S_,
                                            (long long)S_ * DL_, (long long)S_ * DL_,
                                            (long long)S_ * S_, 0.125f);
    }

    // 5) row stats (LN1) + est_scale
    row_stats_est<<<H_ * S_, 256, 0, stream>>>(attn, scw, scb, mnb, rsb, estb);

    // 6) fused LN1 + 3-layer causal depthwise conv (TDM tile ingest, LDS ping-pong)
    {
        size_t shmem = (size_t)2 * S_ * LST_ * sizeof(float);
        (void)hipFuncSetAttribute((const void*)cnn_kernel,
                                  hipFuncAttributeMaxDynamicSharedMemorySize,
                                  (int)shmem);
        cnn_kernel<<<dim3(S_ / KT_, H_), 256, shmem, stream>>>(attn, convw, convb,
                                                               mnb, rsb, ln1w, ln1b);
    }

    // 7) LN2 + mask + softmax + sigmoid scale -> bf16 p
    softmax_kernel<<<H_ * S_, 256, 0, stream>>>(attn, mask, ln2w, ln2b, estb, pb);

    // 8) context: per head [1024,1024] x [128,1024]^T -> ctx[q, h*128+d]
    {
        dim3 g(DH_ / 64, S_ / 128, H_), b(256);
        gemm_bf16_wmma<<<g, b, 0, stream>>>(pb, vbt, ctx + 0, S_, S_, S_, HID_,
                                            (long long)S_ * S_, (long long)DH_ * S_,
                                            (long long)DH_, 1.0f);
    }

    // 9) output projection: [1024,4096] x [4096,4096]^T
    f32_to_bf16_kernel<<<2048, 256, 0, stream>>>(ctx, ctxb, (int)nXH);
    {
        dim3 g(HID_ / 64, S_ / 128, 1), b(256);
        gemm_bf16_wmma<<<g, b, 0, stream>>>(ctxb, Wob, out, HID_, HID_, HID_, HID_, 0, 0, 0, 1.0f);
    }
}
